// ConvolutionKAN_8521215115625
// MI455X (gfx1250) — matrix operand order repaired and checked by reading, not compiled, not run on hardware
//
#include <hip/hip_runtime.h>
#include <hip/hip_bf16.h>
#include <math.h>

// ---------------------------------------------------------------------------
// ConvolutionKAN on gfx1250: fused B-spline feature expansion + WMMA GEMM.
//   M = 16*62*62 = 61504 = 961 * 64 patch rows
//   K = 288*8 (spline bases) + 288 (silu) = 2592 = 81 chunks of 32
//   N = 128 filters
// Pipelined: double-buffered LDS A tile, prefetched A/B fragments,
// 8 waves = 2(M) x 4(N), each wave owns a 32x32 output tile (4 WMMAs/chunk).
// ---------------------------------------------------------------------------

typedef _Float16 v16h __attribute__((ext_vector_type(16)));
typedef _Float16 v8h  __attribute__((ext_vector_type(8)));
typedef float    v8f  __attribute__((ext_vector_type(8)));

#define KCHUNKS      81      // 72 spline chunks + 9 silu chunks
#define SPLINE_CHNK  72
#define WG_COUNT     961     // 61504 / 64
#define LDS_STRIDE   40      // halves per A-tile row (80B, 16B aligned, bank-spread)

// Uniform grid: g[j] = -1 + 0.4*(j-3), j = 0..11 (matches make_grid bitwise)
__device__ __forceinline__ float kan_grid(int j) {
    return -1.0f + 0.4f * (float)(j - 3);
}

// Cubic Cox-de Boor on the uniform grid -> 8 basis values (as f16 vector)
__device__ __forceinline__ v8h bspline8(float x) {
    float b[11];
#pragma unroll
    for (int j = 0; j < 11; ++j) {
        float g0 = kan_grid(j), g1 = kan_grid(j + 1);
        b[j] = (x >= g0 && x < g1) ? 1.0f : 0.0f;
    }
#pragma unroll
    for (int k = 1; k <= 3; ++k) {
        float inv = 1.0f / (0.4f * (float)k);
#pragma unroll
        for (int j = 0; j <= 10 - k; ++j) {
            float gj   = kan_grid(j);
            float gjk1 = kan_grid(j + k + 1);
            b[j] = (x - gj) * inv * b[j] + (gjk1 - x) * inv * b[j + 1];
        }
    }
    v8h o;
#pragma unroll
    for (int j = 0; j < 8; ++j) o[j] = (_Float16)b[j];
    return o;
}

__device__ __forceinline__ float silu_f(float x) {
    return x / (1.0f + expf(-x));
}

// Decompose patch-row index m -> flat element offset of its (b, oh, ow, 0)
__device__ __forceinline__ int row_offset(int m) {
    int bi  = m / 3844;           // 62*62
    int rem = m - bi * 3844;
    int oh  = rem / 62;
    int ow  = rem - oh * 62;
    return ((bi * 64 + oh) * 64 + ow) * 32;   // into inputs[16][64][64][32]
}

// ---------------------------------------------------------------------------
// Prep: fold scale into weights, f16-convert, pre-swizzle to WMMA-B layout:
//   Bpk[c][n][k], k contiguous (K-contiguous column vectors per chunk).
// ---------------------------------------------------------------------------
__global__ __launch_bounds__(256) void kan_prep(const float* __restrict__ spline_kernel, // (288,8,128)
                                                const float* __restrict__ scale,          // (288,128)
                                                _Float16* __restrict__ Bpk) {             // (81,128,32)
    int idx = blockIdx.x * 256 + threadIdx.x;
    if (idx >= KCHUNKS * 128 * 32) return;
    int k = idx & 31;
    int n = (idx >> 5) & 127;
    int c = idx >> 12;
    float v;
    if (c < SPLINE_CHNK) {
        int i = c * 4 + (k >> 3);     // input dim
        int b = k & 7;                // basis index
        v = spline_kernel[(i * 8 + b) * 128 + n] * scale[i * 128 + n];
    } else {
        int i = (c - SPLINE_CHNK) * 32 + k;
        v = scale[i * 128 + n];
    }
    Bpk[idx] = (_Float16)v;
}

// ---- A fragment (16x32 f16): lane = row m, half h holds
//      K = {h*8..h*8+7} U {16+h*8..16+h*8+7} -> two 16B LDS loads ----
__device__ __forceinline__ v16h load_afrag(const _Float16* buf, int mrow, int h) {
    const _Float16* ap = buf + mrow * LDS_STRIDE;
    v8h lo = *(const v8h*)(ap + h * 8);
    v8h hi = *(const v8h*)(ap + 16 + h * 8);
    v16h a;
#pragma unroll
    for (int j = 0; j < 8; ++j) { a[j] = lo[j]; a[j + 8] = hi[j]; }
    return a;
}

// ---- B fragment (32x16 f16): lane = col n, half h holds
//      K = {h*8..h*8+7} U {16+h*8..16+h*8+7} -> two 16B global loads (L2-resident) ----
__device__ __forceinline__ v16h load_bfrag(const _Float16* __restrict__ Bpk,
                                           int c, int ncol, int h) {
    const _Float16* bp = Bpk + ((size_t)c * 128 + ncol) * 32 + h * 8;
    v8h lo = *(const v8h*)(bp);
    v8h hi = *(const v8h*)(bp + 16);
    v16h b;
#pragma unroll
    for (int j = 0; j < 8; ++j) { b[j] = lo[j]; b[j + 8] = hi[j]; }
    return b;
}

// ---------------------------------------------------------------------------
// Main: one WG = 64(M) x 128(N) output tile. 8 waves = 2(M) x 4(N).
// ---------------------------------------------------------------------------
__global__ __launch_bounds__(256) void kan_main(const float*    __restrict__ x,    // (16,64,64,32)
                                                const _Float16* __restrict__ Bpk,  // (81,128,32)
                                                const float*    __restrict__ bias, // (128)
                                                float*          __restrict__ out)  // (61504,128)
{
    __shared__ _Float16 At[2][64 * LDS_STRIDE];   // double-buffered 64x32 f16 A tile

    const int tid  = threadIdx.x;
    const int lane = tid & 31;
    const int wid  = tid >> 5;          // 0..7
    const int mh   = wid & 1;           // M half: rows mh*32..mh*32+31
    const int nq   = wid >> 1;          // N quarter: cols nq*32..nq*32+31
    const int m0   = blockIdx.x * 64;
    const int h    = lane >> 4;         // half-wave
    const int ln   = lane & 15;

    // A-tile fill roles (invariant across chunks)
    const int r_s = tid >> 2;           // spline fill: row 0..63
    const int q_s = tid & 3;            // spline fill: dim-in-chunk 0..3
    const int roff_s = row_offset(m0 + r_s);
    int roff_u[8];
#pragma unroll
    for (int u = 0; u < 8; ++u)
        roff_u[u] = row_offset(m0 + (tid >> 5) * 8 + u);

    // Fragment coordinates for this wave
    const int mrow0 = mh * 32 + ln;
    const int mrow1 = mh * 32 + 16 + ln;
    const int ncol0 = nq * 32 + ln;
    const int ncol1 = nq * 32 + 16 + ln;

    // Bias (epilogue) hoisted
    float bia0 = bias[ncol0];
    float bia1 = bias[ncol1];

    // -------- A-tile fill for chunk c into buf --------
    auto fill_tile = [&](int c, _Float16* buf) {
        if (c < SPLINE_CHNK) {
            // 64 rows x 4 dims, one (row,dim) per thread -> 8 bases each
            int i  = c * 4 + q_s;
            int di = i / 96, dj = (i / 32) % 3, ch = i & 31;
            float xv = x[roff_s + (di * 64 + dj) * 32 + ch];
            *(v8h*)(&buf[r_s * LDS_STRIDE + q_s * 8]) = bspline8(xv);
        } else {
            // 64 rows x 32 dims of silu(x), 8 rows per thread
            int s  = c - SPLINE_CHNK;
            int i  = s * 32 + (tid & 31);
            int di = i / 96, dj = (i / 32) % 3, ch = i & 31;
            int eo = (di * 64 + dj) * 32 + ch;
#pragma unroll
            for (int u = 0; u < 8; ++u) {
                int r = (tid >> 5) * 8 + u;
                float xv = x[roff_u[u] + eo];
                buf[r * LDS_STRIDE + (tid & 31)] = (_Float16)silu_f(xv);
            }
        }
    };

    v8f acc[2][2] = {};
    v16h af[2], bf[2];

    // -------- prologue: fill chunk 0, preload its fragments --------
    fill_tile(0, At[0]);
    __syncthreads();
    af[0] = load_afrag(At[0], mrow0, h);
    af[1] = load_afrag(At[0], mrow1, h);
    bf[0] = load_bfrag(Bpk, 0, ncol0, h);
    bf[1] = load_bfrag(Bpk, 0, ncol1, h);

    // -------- pipelined K loop: one barrier per chunk --------
    for (int c = 0; c < KCHUNKS; ++c) {
        const int nxt = c + 1;
        _Float16* nbuf = At[nxt & 1];
        if (nxt < KCHUNKS) fill_tile(nxt, nbuf);   // overlaps WMMAs below

#pragma unroll
        for (int i = 0; i < 2; ++i)
#pragma unroll
            for (int j = 0; j < 2; ++j)
                acc[i][j] = __builtin_amdgcn_wmma_f32_16x16x32_f16(
                    false, af[i], false, bf[j], (short)0, acc[i][j], false, false);

        __syncthreads();                           // fill(nxt) complete everywhere
        if (nxt < KCHUNKS) {
            // prefetch next chunk's fragments; consumed a full iteration later
            af[0] = load_afrag(nbuf, mrow0, h);
            af[1] = load_afrag(nbuf, mrow1, h);
            bf[0] = load_bfrag(Bpk, nxt, ncol0, h);
            bf[1] = load_bfrag(Bpk, nxt, ncol1, h);
        }
    }

    // ---- Epilogue: + bias, store fp32. D layout: (M = h*8 + r, N = ln) ----
#pragma unroll
    for (int i = 0; i < 2; ++i) {
        const size_t mb = (size_t)(m0 + mh * 32 + i * 16 + h * 8);
#pragma unroll
        for (int r = 0; r < 8; ++r) {
            out[(mb + r) * 128 + ncol0] = acc[i][0][r] + bia0;
            out[(mb + r) * 128 + ncol1] = acc[i][1][r] + bia1;
        }
    }
}

extern "C" void kernel_launch(void* const* d_in, const int* in_sizes, int n_in,
                              void* d_out, int out_size, void* d_ws, size_t ws_size,
                              hipStream_t stream) {
    const float* x    = (const float*)d_in[0];   // inputs (16,64,64,32)
    const float* sk   = (const float*)d_in[1];   // spline_kernel (288,8,128)
    const float* sc   = (const float*)d_in[2];   // scale_factor (288,128)
    const float* bias = (const float*)d_in[3];   // bias (128)
    // d_in[4] = grid: uniform, baked into kan_grid()

    _Float16* Bpk = (_Float16*)d_ws;             // 81*128*32 f16 = 663,552 B

    const int prep_elems = KCHUNKS * 128 * 32;
    kan_prep<<<(prep_elems + 255) / 256, 256, 0, stream>>>(sk, sc, Bpk);
    kan_main<<<WG_COUNT, 256, 0, stream>>>(x, Bpk, bias, (float*)d_out);
}
